// TransformerEncoder_50611894616493
// MI455X (gfx1250) — matrix operand order repaired and checked by reading, compiled, not run on hardware
//
#include <hip/hip_runtime.h>
#include <stdint.h>

// ---------------------------------------------------------------------------
// Feature probes (compile-safe: fall back to manual register staging)
// ---------------------------------------------------------------------------
#ifndef __has_builtin
#define __has_builtin(x) 0
#endif

#if __has_builtin(__builtin_amdgcn_global_load_async_to_lds_b128)
#define ASYNC_LDS 1
#else
#define ASYNC_LDS 0
#endif

#if ASYNC_LDS
// Diagnosed signature: (int4 AS1*, int4 AS3*, imm offset, imm cpol)
typedef int v4i_async __attribute__((ext_vector_type(4)));
typedef __attribute__((address_space(1))) v4i_async* gas_ptr_t;
typedef __attribute__((address_space(3))) v4i_async* las_ptr_t;
__device__ __forceinline__ void wait_async_all() {
#if __has_builtin(__builtin_amdgcn_s_wait_asynccnt)
    __builtin_amdgcn_s_wait_asynccnt(0);
#else
    asm volatile("s_wait_asynccnt 0x0" ::: "memory");
#endif
}
#endif

// ---------------------------------------------------------------------------
// Types for CDNA5 WMMA (wave32): 16x16x32 bf16 -> f32 accum
// ---------------------------------------------------------------------------
typedef __attribute__((ext_vector_type(16))) __bf16 v16bf;
typedef __attribute__((ext_vector_type(8)))  float  v8f;

union FragBF16 {
    uint32_t u[8];
    v16bf    v;
};

__device__ __forceinline__ uint16_t f32_to_bf16(float f) {
    uint32_t u = __builtin_bit_cast(uint32_t, f);
    u += 0x7FFFu + ((u >> 16) & 1u);          // round to nearest even
    return (uint16_t)(u >> 16);
}

__device__ __forceinline__ float gelu_exact(float x) {
    return 0.5f * x * (1.0f + erff(x * 0.70710678118654752440f));
}

// ---------------------------------------------------------------------------
// fp32 [D,D] -> bf16 transposed [N,K] (weights), 32x32 tiles via LDS
// ---------------------------------------------------------------------------
__global__ __launch_bounds__(256)
void cvt_transpose_f32_to_bf16(const float* __restrict__ s,
                               uint16_t* __restrict__ d) {
    __shared__ float tile[32][33];
    const int bx = blockIdx.x * 32;   // source col block
    const int by = blockIdx.y * 32;   // source row block
    const int tx = threadIdx.x & 31;
    const int ty = threadIdx.x >> 5;  // 8 rows per pass
    #pragma unroll
    for (int i = 0; i < 32; i += 8)
        tile[ty + i][tx] = s[(size_t)(by + ty + i) * 1024 + bx + tx];
    __syncthreads();
    #pragma unroll
    for (int i = 0; i < 32; i += 8)
        d[(size_t)(bx + ty + i) * 1024 + by + tx] = f32_to_bf16(tile[tx][ty + i]);
}

// ---------------------------------------------------------------------------
// LayerNorm over D=1024, one block per row, writes bf16
// ---------------------------------------------------------------------------
__global__ __launch_bounds__(256)
void layernorm_to_bf16(const float* __restrict__ x, const float* __restrict__ g,
                       const float* __restrict__ b, uint16_t* __restrict__ out) {
    const int row = blockIdx.x;
    const int t = threadIdx.x;
    const float* xr = x + (size_t)row * 1024;
    float s = 0.f, s2 = 0.f;
    #pragma unroll
    for (int j = 0; j < 4; ++j) {
        float v = xr[t + j * 256];
        s += v; s2 += v * v;
    }
    __shared__ float r1[256], r2[256];
    r1[t] = s; r2[t] = s2;
    __syncthreads();
    for (int off = 128; off > 0; off >>= 1) {
        if (t < off) { r1[t] += r1[t + off]; r2[t] += r2[t + off]; }
        __syncthreads();
    }
    const float mu   = r1[0] * (1.0f / 1024.0f);
    const float var  = r2[0] * (1.0f / 1024.0f) - mu * mu;
    const float rstd = rsqrtf(var + 1e-5f);
    uint16_t* orow = out + (size_t)row * 1024;
    #pragma unroll
    for (int j = 0; j < 4; ++j) {
        int i = t + j * 256;
        orow[i] = f32_to_bf16((xr[i] - mu) * rstd * g[i] + b[i]);
    }
}

// ---------------------------------------------------------------------------
// Row softmax over 1024 cols, fp32 in -> bf16 probabilities out
// ---------------------------------------------------------------------------
__global__ __launch_bounds__(256)
void softmax_to_bf16(const float* __restrict__ scores,
                     uint16_t* __restrict__ probs) {
    const int row = blockIdx.x;
    const int t = threadIdx.x;
    const float* sr = scores + (size_t)row * 1024;
    float m = -3.4e38f;
    #pragma unroll
    for (int j = 0; j < 4; ++j) m = fmaxf(m, sr[t + j * 256]);
    __shared__ float red[256];
    red[t] = m;
    __syncthreads();
    for (int off = 128; off > 0; off >>= 1) {
        if (t < off) red[t] = fmaxf(red[t], red[t + off]);
        __syncthreads();
    }
    m = red[0];
    __syncthreads();
    float e[4], s = 0.f;
    #pragma unroll
    for (int j = 0; j < 4; ++j) { e[j] = expf(sr[t + j * 256] - m); s += e[j]; }
    red[t] = s;
    __syncthreads();
    for (int off = 128; off > 0; off >>= 1) {
        if (t < off) red[t] += red[t + off];
        __syncthreads();
    }
    const float inv = 1.0f / red[0];
    uint16_t* pr = probs + (size_t)row * 1024;
    #pragma unroll
    for (int j = 0; j < 4; ++j) pr[t + j * 256] = f32_to_bf16(e[j] * inv);
}

// ---------------------------------------------------------------------------
// Double-buffered tiled bf16 WMMA GEMM.
//   C[M,N] = epilogue(A[M,K] * op(B) * scale + bias [+ resid])
// Block tile 128x128, 8 waves, each wave 64x32 -> 4x2 wmma frags.
// TRANSB: B stored [N,K] row-major (all weight GEMMs + Q*K^T) -> contiguous
//         128-bit tile copies staged via GLOBAL_LOAD_ASYNC_TO_LDS_B128.
// !TRANSB: B stored [K,N] (attn @ v) -> bf16 pair packing in registers.
// EPI: 0 bf16(+bias); 1 f32 scaled; 2 bf16 gelu(+bias);
//      3 f32 +bias+resid; 4 f32 gelu(+bias)+resid
// Dims are multiples of 128 -> no bounds checks, EXEC all-1s around WMMA.
// ---------------------------------------------------------------------------
template <int EPI, bool TRANSB>
__global__ __launch_bounds__(256)
void gemm_bf16_wmma(const uint16_t* __restrict__ A, long long sA,
                    const uint16_t* __restrict__ B, long long sB,
                    const float* __restrict__ bias,
                    const float* __restrict__ resid, long long sR,
                    void* __restrict__ Cout, long long sC,
                    int lda, int ldb, int ldc, int K, float scale) {
    // Double-buffered LDS tiles, padded to 17 dwords/row (bank-conflict free).
    // Asu[m][kp] packs A[m][2kp..2kp+1]; Bsu[n][kp] packs B[2kp..2kp+1][n].
    __shared__ uint32_t Asu[2][128][17];
    __shared__ uint32_t Bsu[2][128][17];

    const int batch = blockIdx.z;
    A += (size_t)batch * sA;
    B += (size_t)batch * sB;
    const int rowBase = blockIdx.y * 128;
    const int colBase = blockIdx.x * 128;

    const int t    = threadIdx.x;
    const int lane = t & 31;
    const int wave = t >> 5;
    const int wm   = wave >> 2;   // 0..1 : 64-row wave tile
    const int wn   = wave & 3;    // 0..3 : 32-col wave tile
    const int half = lane >> 4;
    const int r    = lane & 15;

    // copy-role indices (A / transposed-B: 64B rows in 16B chunks)
    const int crow = t >> 2;      // +64*i
    const int cq   = t & 3;
    // NN-B pack roles
    const int cn   = t & 127;
    const int ckp  = t >> 7;      // +2*i

    v8f acc[4][2];
    {
        const v8f vz = {0.f, 0.f, 0.f, 0.f, 0.f, 0.f, 0.f, 0.f};
        #pragma unroll
        for (int mi = 0; mi < 4; ++mi)
            #pragma unroll
            for (int ni = 0; ni < 2; ++ni) acc[mi][ni] = vz;
    }

    uint4 aReg[2];
    uint4 bReg[2];
    uint32_t bPack[8];

    // issue global traffic for tile at k0 (dest buffer `buf`)
    auto stageIssue = [&](int k0, int buf) {
#if ASYNC_LDS
        #pragma unroll
        for (int i = 0; i < 2; ++i) {
            const int row = crow + 64 * i;
            __builtin_amdgcn_global_load_async_to_lds_b128(
                (gas_ptr_t)(A + (size_t)(rowBase + row) * lda + k0 + cq * 8),
                (las_ptr_t)&Asu[buf][row][cq * 4], 0, 0);
        }
        if (TRANSB) {
            #pragma unroll
            for (int i = 0; i < 2; ++i) {
                const int row = crow + 64 * i;
                __builtin_amdgcn_global_load_async_to_lds_b128(
                    (gas_ptr_t)(B + (size_t)(colBase + row) * ldb + k0 + cq * 8),
                    (las_ptr_t)&Bsu[buf][row][cq * 4], 0, 0);
            }
        } else {
            #pragma unroll
            for (int i = 0; i < 8; ++i) {
                const int kp = ckp + 2 * i;
                uint32_t lo = B[(size_t)(k0 + 2 * kp) * ldb + colBase + cn];
                uint32_t hi = B[(size_t)(k0 + 2 * kp + 1) * ldb + colBase + cn];
                bPack[i] = lo | (hi << 16);
            }
        }
#else
        #pragma unroll
        for (int i = 0; i < 2; ++i)
            aReg[i] = *(const uint4*)(A + (size_t)(rowBase + crow + 64 * i) * lda +
                                      k0 + cq * 8);
        if (TRANSB) {
            #pragma unroll
            for (int i = 0; i < 2; ++i)
                bReg[i] = *(const uint4*)(B + (size_t)(colBase + crow + 64 * i) * ldb +
                                          k0 + cq * 8);
        } else {
            #pragma unroll
            for (int i = 0; i < 8; ++i) {
                const int kp = ckp + 2 * i;
                uint32_t lo = B[(size_t)(k0 + 2 * kp) * ldb + colBase + cn];
                uint32_t hi = B[(size_t)(k0 + 2 * kp + 1) * ldb + colBase + cn];
                bPack[i] = lo | (hi << 16);
            }
        }
#endif
    };

    // flush register-staged data into LDS buffer `buf`
    auto stageFlush = [&](int buf) {
#if ASYNC_LDS
        if (!TRANSB) {
            #pragma unroll
            for (int i = 0; i < 8; ++i) Bsu[buf][cn][ckp + 2 * i] = bPack[i];
        }
#else
        #pragma unroll
        for (int i = 0; i < 2; ++i) {
            const int row = crow + 64 * i;
            Asu[buf][row][cq * 4 + 0] = aReg[i].x;
            Asu[buf][row][cq * 4 + 1] = aReg[i].y;
            Asu[buf][row][cq * 4 + 2] = aReg[i].z;
            Asu[buf][row][cq * 4 + 3] = aReg[i].w;
        }
        if (TRANSB) {
            #pragma unroll
            for (int i = 0; i < 2; ++i) {
                const int row = crow + 64 * i;
                Bsu[buf][row][cq * 4 + 0] = bReg[i].x;
                Bsu[buf][row][cq * 4 + 1] = bReg[i].y;
                Bsu[buf][row][cq * 4 + 2] = bReg[i].z;
                Bsu[buf][row][cq * 4 + 3] = bReg[i].w;
            }
        } else {
            #pragma unroll
            for (int i = 0; i < 8; ++i) Bsu[buf][cn][ckp + 2 * i] = bPack[i];
        }
#endif
    };

    // Load ALL fragments first, then fire the 8 WMMAs back-to-back so the
    // DS waits are hoisted ahead of the whole XDL burst.
    auto computeTile = [&](int cur) {
        FragBF16 bfrag[2];
        FragBF16 afrag[4];
        #pragma unroll
        for (int ni = 0; ni < 2; ++ni)
            #pragma unroll
            for (int v = 0; v < 8; ++v)
                bfrag[ni].u[v] = Bsu[cur][wn * 32 + ni * 16 + r][half * 8 + v];
        #pragma unroll
        for (int mi = 0; mi < 4; ++mi)
            #pragma unroll
            for (int v = 0; v < 8; ++v) {
                const int kp = half * 8 + v;  // contiguous k = 16*half + s (matches B)
                afrag[mi].u[v] = Asu[cur][wm * 64 + mi * 16 + r][kp];
            }
        #pragma unroll
        for (int mi = 0; mi < 4; ++mi)
            #pragma unroll
            for (int ni = 0; ni < 2; ++ni)
                acc[mi][ni] = __builtin_amdgcn_wmma_f32_16x16x32_bf16(
                    false, afrag[mi].v, false, bfrag[ni].v, (short)0,
                    acc[mi][ni], false, false);
    };

    const int NK = K >> 5;

    // ---- software pipeline: prefetch tile kt+1 while computing tile kt ----
    stageIssue(0, 0);
    stageFlush(0);
    for (int kt = 0; kt < NK; ++kt) {
        const int cur = kt & 1;
#if ASYNC_LDS
        wait_async_all();
#endif
        __syncthreads();
        const bool more = (kt + 1 < NK);
        if (more) stageIssue((kt + 1) << 5, cur ^ 1);
        if (kt + 2 < NK)
            __builtin_prefetch(A + (size_t)(rowBase + crow) * lda + ((kt + 2) << 5),
                               0, 3);
        computeTile(cur);
        if (more) stageFlush(cur ^ 1);
    }

    // ---- fused epilogue ----
    #pragma unroll
    for (int mi = 0; mi < 4; ++mi) {
        #pragma unroll
        for (int ni = 0; ni < 2; ++ni) {
            const int col = colBase + wn * 32 + ni * 16 + r;
            const float bv = bias ? bias[col] : 0.0f;
            #pragma unroll
            for (int v = 0; v < 8; ++v) {
                const int row = rowBase + wm * 64 + mi * 16 + v + 8 * half;
                const size_t off = (size_t)batch * sC + (size_t)row * ldc + col;
                float pre = acc[mi][ni][v] * scale + bv;
                if (EPI == 0) {
                    ((uint16_t*)Cout)[off] = f32_to_bf16(pre);
                } else if (EPI == 1) {
                    ((float*)Cout)[off] = pre;
                } else if (EPI == 2) {
                    ((uint16_t*)Cout)[off] = f32_to_bf16(gelu_exact(pre));
                } else if (EPI == 3) {
                    ((float*)Cout)[off] =
                        pre + resid[(size_t)batch * sR + (size_t)row * ldc + col];
                } else {  // EPI == 4
                    ((float*)Cout)[off] =
                        gelu_exact(pre) +
                        resid[(size_t)batch * sR + (size_t)row * ldc + col];
                }
            }
        }
    }
}

// ---------------------------------------------------------------------------
// Host-side orchestration
// ---------------------------------------------------------------------------
extern "C" void kernel_launch(void* const* d_in, const int* in_sizes, int n_in,
                              void* d_out, int out_size, void* d_ws,
                              size_t ws_size, hipStream_t stream) {
    (void)in_sizes; (void)n_in; (void)out_size; (void)ws_size;

    const int D = 1024, Bb = 16, N = 1024;
    const int M = Bb * N;   // 16384 rows

    const float* x    = (const float*)d_in[0];
    const float* ln1g = (const float*)d_in[1];
    const float* ln1b = (const float*)d_in[2];
    const float* ln2g = (const float*)d_in[3];
    const float* ln2b = (const float*)d_in[4];
    const float* wq = (const float*)d_in[5];  const float* bq = (const float*)d_in[6];
    const float* wk = (const float*)d_in[7];  const float* bk = (const float*)d_in[8];
    const float* wv = (const float*)d_in[9];  const float* bv = (const float*)d_in[10];
    const float* wo = (const float*)d_in[11]; const float* bo = (const float*)d_in[12];
    const float* w1 = (const float*)d_in[13]; const float* b1 = (const float*)d_in[14];
    const float* w2 = (const float*)d_in[15]; const float* b2 = (const float*)d_in[16];

    // Workspace layout; buffers reused once producer-consumer windows close
    // (single stream -> ordering is guaranteed).
    char* ws = (char*)d_ws;
    const size_t MB = 1024ull * 1024ull;
    uint16_t* wqb = (uint16_t*)(ws + 0 * MB);    // transposed bf16 weights, 2 MB each
    uint16_t* wkb = (uint16_t*)(ws + 2 * MB);
    uint16_t* wvb = (uint16_t*)(ws + 4 * MB);
    uint16_t* wob = (uint16_t*)(ws + 6 * MB);
    uint16_t* w1b = (uint16_t*)(ws + 8 * MB);
    uint16_t* w2b = (uint16_t*)(ws + 10 * MB);
    uint16_t* xn  = (uint16_t*)(ws + 12 * MB);   // 32 MB; reused for probs
    uint16_t* qb  = (uint16_t*)(ws + 44 * MB);   // 32 MB; reused for attn_out
    uint16_t* kb  = (uint16_t*)(ws + 76 * MB);   // 32 MB; reused for xu (bf16)
    uint16_t* vb  = (uint16_t*)(ws + 108 * MB);  // 32 MB; reused for h (bf16)
    float*    sc  = (float*)(ws + 140 * MB);     // 64 MB scores
    float*    xup = (float*)(ws + 204 * MB);     // 64 MB x_update (fp32)
    float*    out = (float*)d_out;

    // 1) weights fp32 -> bf16, transposed to [N,K] so every weight GEMM is NT
    const dim3 gT(32, 32);
    cvt_transpose_f32_to_bf16<<<gT, 256, 0, stream>>>(wq, wqb);
    cvt_transpose_f32_to_bf16<<<gT, 256, 0, stream>>>(wk, wkb);
    cvt_transpose_f32_to_bf16<<<gT, 256, 0, stream>>>(wv, wvb);
    cvt_transpose_f32_to_bf16<<<gT, 256, 0, stream>>>(wo, wob);
    cvt_transpose_f32_to_bf16<<<gT, 256, 0, stream>>>(w1, w1b);
    cvt_transpose_f32_to_bf16<<<gT, 256, 0, stream>>>(w2, w2b);

    // 2) xn = LN1(x) -> bf16
    layernorm_to_bf16<<<M, 256, 0, stream>>>(x, ln1g, ln1b, xn);

    const dim3 blk(256);
    const dim3 gProj(D / 128, M / 128, 1);   // (8,128,1)
    const dim3 gAttn(N / 128, N / 128, Bb);  // (8,8,16)
    const long long sND = (long long)N * D;
    const long long sNN = (long long)N * N;

    // 3) Q,K,V projections (bf16 out + bias), NT
    gemm_bf16_wmma<0, true><<<gProj, blk, 0, stream>>>(
        xn, 0, wqb, 0, bq, nullptr, 0, qb, 0, D, D, D, D, 1.0f);
    gemm_bf16_wmma<0, true><<<gProj, blk, 0, stream>>>(
        xn, 0, wkb, 0, bk, nullptr, 0, kb, 0, D, D, D, D, 1.0f);
    gemm_bf16_wmma<0, true><<<gProj, blk, 0, stream>>>(
        xn, 0, wvb, 0, bv, nullptr, 0, vb, 0, D, D, D, D, 1.0f);

    // 4) scores[b] = (q[b] @ k[b]^T) / sqrt(D), fp32 out
    gemm_bf16_wmma<1, true><<<gAttn, blk, 0, stream>>>(
        qb, sND, kb, sND, nullptr, nullptr, 0, sc, sNN, D, D, N, D,
        0.03125f /* 1/sqrt(1024) */);

    // 5) probs = softmax(scores) -> bf16 (reuses xn)
    softmax_to_bf16<<<M, 256, 0, stream>>>(sc, xn);

    // 6) attn_out[b] = probs[b] @ v[b] -> bf16 (reuses qb), NN
    gemm_bf16_wmma<0, false><<<gAttn, blk, 0, stream>>>(
        xn, sNN, vb, sND, nullptr, nullptr, 0, qb, sND, N, D, D, N, 1.0f);

    // 7) x_update = attn_out @ wo + bo + x, fp32
    gemm_bf16_wmma<3, true><<<gProj, blk, 0, stream>>>(
        qb, 0, wob, 0, bo, x, 0, xup, 0, D, D, D, D, 1.0f);

    // 8) xu = LN2(x_update) -> bf16 (reuses kb)
    layernorm_to_bf16<<<M, 256, 0, stream>>>(xup, ln2g, ln2b, kb);

    // 9) h = gelu(xu @ w1 + b1) -> bf16 (reuses vb)
    gemm_bf16_wmma<2, true><<<gProj, blk, 0, stream>>>(
        kb, 0, w1b, 0, b1, nullptr, 0, vb, 0, D, D, D, D, 1.0f);

    // 10) out = gelu(h @ w2 + b2) + x_update, fp32 to d_out
    gemm_bf16_wmma<4, true><<<gProj, blk, 0, stream>>>(
        vb, 0, w2b, 0, b2, xup, 0, out, 0, D, D, D, D, 1.0f);
}
